// PatchcoreModel_5901285065146
// MI455X (gfx1250) — matrix operand order repaired and checked by reading, not compiled, not run on hardware
//
#include <hip/hip_runtime.h>
#include <hip/hip_bf16.h>
#include <math.h>

typedef __bf16 bf16;
typedef __bf16 v16bf __attribute__((ext_vector_type(16)));
typedef __bf16 v8bf  __attribute__((ext_vector_type(8)));
typedef float  v8f   __attribute__((ext_vector_type(8)));

#define BATCH 8
#define C2    512
#define C3    1024
#define EDIM  1536
#define HW    28
#define PPB   784      // 28*28 patches per batch image
#define PTOT  6272     // BATCH*PPB
#define MROWS 20000
#define H3    14
#define OUTSZ 224
#define KSZ   33
#define KRAD  16
#define SIGMA 4.0f
#define NTILES_M 1250  // 20000/16

#if defined(__has_builtin)
#if __has_builtin(__builtin_amdgcn_global_load_async_to_lds_b128)
#define HAVE_ASYNC_LDS 1
#endif
#endif

#if HAVE_ASYNC_LDS
// exact param types per hipcc diagnostic: v4i (vector_size 16) in AS1 / AS3
typedef int v4i_t __attribute__((vector_size(16)));
typedef __attribute__((address_space(1))) v4i_t g_v4i;
typedef __attribute__((address_space(3))) v4i_t l_v4i;
#endif

__device__ __forceinline__ int iclampi(int v, int lo, int hi) {
  return v < lo ? lo : (v > hi ? hi : v);
}

// ---------------- Kernel 1a: 3x3 avg pool (zero pad, /9) on feat3 ----------------
__global__ __launch_bounds__(256) void k_pool3(const float* __restrict__ f3,
                                               float* __restrict__ f3p) {
  int idx = blockIdx.x * 256 + threadIdx.x;      // BATCH*C3*14*14 = 1605632 exact
  int x = idx % H3;
  int y = (idx / H3) % H3;
  int c = idx / (H3 * H3);                       // combined b*C3 + c
  const float* base = f3 + (size_t)c * (H3 * H3);
  float s = 0.0f;
  for (int dy = -1; dy <= 1; ++dy) {
    int yy = y + dy; if ((unsigned)yy > (unsigned)(H3 - 1)) continue;
    for (int dx = -1; dx <= 1; ++dx) {
      int xx = x + dx; if ((unsigned)xx > (unsigned)(H3 - 1)) continue;
      s += base[yy * H3 + xx];
    }
  }
  f3p[idx] = s * (1.0f / 9.0f);
}

// ---------------- Kernel 1b: build embedding [PTOT][EDIM] (f32 + bf16) -----------
__global__ __launch_bounds__(256) void k_embed(const float* __restrict__ f2,
                                               const float* __restrict__ f3p,
                                               float* __restrict__ embF,
                                               bf16* __restrict__ embB) {
  size_t idx = (size_t)blockIdx.x * 256 + threadIdx.x;  // PTOT*EDIM = 9633792 exact
  int e = (int)(idx % EDIM);
  size_t p = idx / EDIM;
  int x = (int)(p % HW);
  int y = (int)((p / HW) % HW);
  int b = (int)(p / PPB);
  float v;
  if (e < C2) {
    const float* base = f2 + ((size_t)(b * C2 + e)) * (HW * HW);
    float s = 0.0f;
    for (int dy = -1; dy <= 1; ++dy) {
      int yy = y + dy; if ((unsigned)yy > (unsigned)(HW - 1)) continue;
      for (int dx = -1; dx <= 1; ++dx) {
        int xx = x + dx; if ((unsigned)xx > (unsigned)(HW - 1)) continue;
        s += base[yy * HW + xx];
      }
    }
    v = s * (1.0f / 9.0f);
  } else {
    int c = e - C2;
    const float* base = f3p + ((size_t)(b * C3 + c)) * (H3 * H3);
    float iy = (y + 0.5f) * 0.5f - 0.5f;
    float ix = (x + 0.5f) * 0.5f - 0.5f;
    int y0 = (int)floorf(iy), x0 = (int)floorf(ix);
    float fy = iy - (float)y0, fx = ix - (float)x0;
    int y0c = iclampi(y0, 0, H3 - 1), y1c = iclampi(y0 + 1, 0, H3 - 1);
    int x0c = iclampi(x0, 0, H3 - 1), x1c = iclampi(x0 + 1, 0, H3 - 1);
    v = (1.0f - fy) * ((1.0f - fx) * base[y0c * H3 + x0c] + fx * base[y0c * H3 + x1c])
      +         fy  * ((1.0f - fx) * base[y1c * H3 + x0c] + fx * base[y1c * H3 + x1c]);
  }
  embF[idx] = v;
  embB[idx] = (bf16)v;
}

// ---------------- Kernel 1c: xnorm[p] = sum_e bf16(emb)^2 ------------------------
__global__ __launch_bounds__(256) void k_xnorm(const bf16* __restrict__ embB,
                                               float* __restrict__ xnorm) {
  __shared__ float sv[256];
  int p = blockIdx.x, t = threadIdx.x;
  float s = 0.0f;
  for (int e = t; e < EDIM; e += 256) {
    float v = (float)embB[(size_t)p * EDIM + e];
    s += v * v;
  }
  sv[t] = s; __syncthreads();
  for (int st = 128; st > 0; st >>= 1) { if (t < st) sv[t] += sv[t + st]; __syncthreads(); }
  if (t == 0) xnorm[p] = sv[0];
}

// ---------------- Kernel 2: memory bank -> bf16 + ynorm --------------------------
__global__ __launch_bounds__(256) void k_mbprep(const float* __restrict__ mb,
                                                bf16* __restrict__ mbB,
                                                float* __restrict__ ynorm) {
  __shared__ float sv[256];
  int m = blockIdx.x, t = threadIdx.x;
  float s = 0.0f;
  for (int e = t; e < EDIM; e += 256) {
    float v = mb[(size_t)m * EDIM + e];
    bf16 bv = (bf16)v;
    float fv = (float)bv;
    mbB[(size_t)m * EDIM + e] = bv;
    s += fv * fv;
  }
  sv[t] = s; __syncthreads();
  for (int st = 128; st > 0; st >>= 1) { if (t < st) sv[t] += sv[t + st]; __syncthreads(); }
  if (t == 0) ynorm[m] = sv[0];
}

// ---------------- Kernel 3: bf16 WMMA GEMM fused with min/argmin -----------------
// grid.x = PTOT/16 = 392 workgroups of 128 threads (4 waves).
// A panel (16 patches x 1536 bf16 = 48KB) staged in LDS (async-to-LDS when
// available). 1250 m-tiles = 312 quads (78/wave, SALU-only control flow via
// readfirstlane) + one tail pair on wave 0. 4 accumulator chains per wave.
template <int NT>
__device__ __forceinline__ void gemm_tiles(const bf16* __restrict__ mbB,
                                           const float* __restrict__ ynorm,
                                           const bf16* arow, int tile0,
                                           int row, int half,
                                           float mv[8], unsigned mi[8]) {
  int m[NT];
  float yn[NT];
  const bf16* brow[NT];
  #pragma unroll
  for (int j = 0; j < NT; ++j) {
    m[j] = (tile0 + j) * 16;
    yn[j] = ynorm[m[j] + row];
    brow[j] = mbB + (size_t)(m[j] + row) * EDIM + half * 8;
  }
  v8f acc[NT];
  #pragma unroll
  for (int j = 0; j < NT; ++j) {
    #pragma unroll
    for (int r = 0; r < 8; ++r) acc[j][r] = 0.0f;
  }
  #pragma unroll 2
  for (int kc = 0; kc < EDIM / 32; ++kc) {
    const int k0 = kc * 32;
    // A fragment: lanes 0-15 -> K {0..7,16..23}+k0 ; lanes 16-31 -> K {8..15,24..31}+k0
    v8bf alo = *(const v8bf*)(arow + k0);
    v8bf ahi = *(const v8bf*)(arow + k0 + 16);
    v16bf A;
    #pragma unroll
    for (int i = 0; i < 8; ++i) { A[i] = alo[i]; A[i + 8] = ahi[i]; }
    // B fragment: lane holds column m+row, K = {half*8+0..7, 16+half*8+0..7}
    #pragma unroll
    for (int j = 0; j < NT; ++j) {
      v8bf blo = *(const v8bf*)(brow[j] + k0);
      v8bf bhi = *(const v8bf*)(brow[j] + k0 + 16);
      v16bf Bj;
      #pragma unroll
      for (int i = 0; i < 8; ++i) { Bj[i] = blo[i]; Bj[i + 8] = bhi[i]; }
      acc[j] = __builtin_amdgcn_wmma_f32_16x16x32_bf16(false, A, false, Bj,
                                                       (short)0, acc[j], false, false);
    }
  }
  // C layout: VGPR r, lane -> patch = half*8 + r, column = row
  #pragma unroll
  for (int j = 0; j < NT; ++j) {
    #pragma unroll
    for (int r = 0; r < 8; ++r) {
      float v = yn[j] - 2.0f * acc[j][r];
      unsigned ii = (unsigned)(m[j] + row);
      if (v < mv[r] || (v == mv[r] && ii < mi[r])) { mv[r] = v; mi[r] = ii; }
    }
  }
}

__global__ __launch_bounds__(128) void k_gemm_min(const bf16* __restrict__ embB,
                                                  const bf16* __restrict__ mbB,
                                                  const float* __restrict__ xnorm,
                                                  const float* __restrict__ ynorm,
                                                  float* __restrict__ pscore,
                                                  int* __restrict__ ploc) {
  __shared__ __align__(16) bf16 ldsA[16 * EDIM];
  __shared__ float    redV[4][16];
  __shared__ unsigned redI[4][16];

  const int tid  = threadIdx.x;
  const int wave = __builtin_amdgcn_readfirstlane(tid >> 5);  // SGPR: scalar branches only
  const int lane = tid & 31;
  const int row  = lane & 15;   // A row / B column within tile
  const int half = lane >> 4;   // lane-half selects K sub-range
  const int p0   = blockIdx.x * 16;

  // cooperative A panel load: 48KB contiguous (3072 x 16B)
  {
    const char* src = (const char*)(embB + (size_t)p0 * EDIM);
    char* dstg = (char*)ldsA;
#if HAVE_ASYNC_LDS
    for (int i = tid; i < 3072; i += 128) {
      g_v4i* g = (g_v4i*)(unsigned long long)(size_t)(src + (size_t)i * 16);
      l_v4i* l = (l_v4i*)(unsigned)(size_t)(dstg + (size_t)i * 16);
      __builtin_amdgcn_global_load_async_to_lds_b128(g, l, 0, 0);
    }
#if __has_builtin(__builtin_amdgcn_s_wait_asynccnt)
    __builtin_amdgcn_s_wait_asynccnt(0);
#else
    asm volatile("s_wait_asynccnt 0" ::: "memory");
#endif
#else
    const uint4* s4 = (const uint4*)src;
    uint4* d4 = (uint4*)dstg;
    #pragma unroll
    for (int i = 0; i < 3072 / 128; ++i)
      d4[tid + i * 128] = s4[tid + i * 128];
#endif
  }
  __syncthreads();

  float    mv[8];
  unsigned mi[8];
  #pragma unroll
  for (int r = 0; r < 8; ++r) { mv[r] = 3.4e38f; mi[r] = 0u; }

  const bf16* arow = ldsA + row * EDIM + half * 8;

  // 312 quads of 4 m-tiles, wave-strided (uniform trip count -> no EXEC games)
  for (int q = wave; q < 312; q += 4) {
    if (q + 4 < 312)   // warm a row of the next quad (global_prefetch_b8)
      __builtin_prefetch(mbB + (size_t)((q + 4) * 64 + row) * EDIM + half * 16, 0, 3);
    gemm_tiles<4>(mbB, ynorm, arow, 4 * q, row, half, mv, mi);
  }
  // tail: tiles 1248,1249 on wave 0 (scalar branch, EXEC stays full)
  if (wave == 0)
    gemm_tiles<2>(mbB, ynorm, arow, 1248, row, half, mv, mi);

  // reduce across the 16 lanes of each half (xor masks 1,2,4,8 stay in-half)
  #pragma unroll
  for (int msk = 1; msk <= 8; msk <<= 1) {
    #pragma unroll
    for (int r = 0; r < 8; ++r) {
      float    ov = __shfl_xor(mv[r], msk, 32);
      unsigned oi = __shfl_xor(mi[r], msk, 32);
      if (ov < mv[r] || (ov == mv[r] && oi < mi[r])) { mv[r] = ov; mi[r] = oi; }
    }
  }
  if (row == 0) {  // lane 0 holds patches 0..7, lane 16 holds patches 8..15
    #pragma unroll
    for (int r = 0; r < 8; ++r) { redV[wave][half * 8 + r] = mv[r]; redI[wave][half * 8 + r] = mi[r]; }
  }
  __syncthreads();
  if (tid < 16) {
    float bv = redV[0][tid]; unsigned bi = redI[0][tid];
    for (int w = 1; w < 4; ++w) {
      float ov = redV[w][tid]; unsigned oi = redI[w][tid];
      if (ov < bv || (ov == bv && oi < bi)) { bv = ov; bi = oi; }
    }
    int p = p0 + tid;
    float d2 = xnorm[p] + bv;
    pscore[p] = sqrtf(fmaxf(d2, 0.0f));
    ploc[p] = (int)bi;
  }
}

// ---------------- Kernel 4: per-batch anomaly score (exact f32) ------------------
__global__ __launch_bounds__(256) void k_score(const float* __restrict__ pscore,
                                               const int* __restrict__ ploc,
                                               const float* __restrict__ embF,
                                               const float* __restrict__ mbF,
                                               float* __restrict__ dbank,
                                               float* __restrict__ out_pred) {
  const int b = blockIdx.x, t = threadIdx.x;
  __shared__ float sv[256];
  __shared__ float sv2[256];
  __shared__ int   si[256];
  __shared__ float feat[EDIM];   // staged nn_sample, later max_feat
  __shared__ float s_score; __shared__ int s_maxp; __shared__ int s_nn;
  __shared__ float s_n2;
  __shared__ int   sel[9];
  __shared__ float sd[9];

  // 1) argmax over this batch's 784 patch scores (first occurrence)
  float bvv = -3.4e38f; int bii = 0;
  for (int j = t; j < PPB; j += 256) {
    float v = pscore[b * PPB + j];
    if (v > bvv || (v == bvv && j < bii)) { bvv = v; bii = j; }
  }
  sv[t] = bvv; si[t] = bii; __syncthreads();
  for (int s = 128; s > 0; s >>= 1) {
    if (t < s) {
      float ov = sv[t + s]; int oi = si[t + s];
      if (ov > sv[t] || (ov == sv[t] && oi < si[t])) { sv[t] = ov; si[t] = oi; }
    }
    __syncthreads();
  }
  if (t == 0) { s_score = sv[0]; s_maxp = si[0]; s_nn = ploc[b * PPB + si[0]]; }
  __syncthreads();
  const int nn = s_nn, maxp = s_maxp;

  // 2) stage nn_sample (f32) and its norm
  float part = 0.0f;
  for (int e = t; e < EDIM; e += 256) {
    float v = mbF[(size_t)nn * EDIM + e]; feat[e] = v; part += v * v;
  }
  sv[t] = part; __syncthreads();
  for (int s = 128; s > 0; s >>= 1) { if (t < s) sv[t] += sv[t + s]; __syncthreads(); }
  if (t == 0) s_n2 = sv[0];
  __syncthreads();
  const float nn2 = s_n2;

  // 3) d_bank[m] = ||nn - mb[m]||^2 (f32 exact; sqrt monotone, not needed)
  for (int m = t; m < MROWS; m += 256) {
    const float* r = mbF + (size_t)m * EDIM;
    float dot = 0.0f, m2 = 0.0f;
    for (int e = 0; e < EDIM; ++e) { float v = r[e]; dot += v * feat[e]; m2 += v * v; }
    dbank[(size_t)b * MROWS + m] = nn2 - 2.0f * dot + m2;
  }
  __syncthreads();

  // 4) 9 smallest (iterative selection, ties -> lower index, like lax.top_k)
  for (int k = 0; k < 9; ++k) {
    float bv2 = 3.4e38f; int bi2 = 0x7fffffff;
    for (int m = t; m < MROWS; m += 256) {
      bool skip = false;
      for (int q = 0; q < k; ++q) if (sel[q] == m) skip = true;
      if (skip) continue;
      float v = dbank[(size_t)b * MROWS + m];
      if (v < bv2 || (v == bv2 && m < bi2)) { bv2 = v; bi2 = m; }
    }
    sv[t] = bv2; si[t] = bi2; __syncthreads();
    for (int s = 128; s > 0; s >>= 1) {
      if (t < s) {
        float ov = sv[t + s]; int oi = si[t + s];
        if (ov < sv[t] || (ov == sv[t] && oi < si[t])) { sv[t] = ov; si[t] = oi; }
      }
      __syncthreads();
    }
    if (t == 0) sel[k] = si[0];
    __syncthreads();
  }

  // 5) stage max_feat (f32 embedding row) and its norm
  float p2 = 0.0f;
  const float* mf = embF + (size_t)(b * PPB + maxp) * EDIM;
  for (int e = t; e < EDIM; e += 256) { float v = mf[e]; feat[e] = v; p2 += v * v; }
  sv[t] = p2; __syncthreads();
  for (int s = 128; s > 0; s >>= 1) { if (t < s) sv[t] += sv[t + s]; __syncthreads(); }
  if (t == 0) s_n2 = sv[0];
  __syncthreads();
  const float mf2 = s_n2;

  // distances max_feat -> each support
  for (int k = 0; k < 9; ++k) {
    const float* r = mbF + (size_t)sel[k] * EDIM;
    float dot = 0.0f, m2 = 0.0f;
    for (int e = t; e < EDIM; e += 256) { float v = r[e]; dot += v * feat[e]; m2 += v * v; }
    sv[t] = dot; sv2[t] = m2; __syncthreads();
    for (int s = 128; s > 0; s >>= 1) {
      if (t < s) { sv[t] += sv[t + s]; sv2[t] += sv2[t + s]; }
      __syncthreads();
    }
    if (t == 0) sd[k] = sqrtf(fmaxf(mf2 - 2.0f * sv[0] + sv2[0], 0.0f));
    __syncthreads();
  }

  // 6) weights = 1 - softmax(d)[0]; pred = weights * score
  if (t == 0) {
    float mx = sd[0];
    for (int k = 1; k < 9; ++k) mx = fmaxf(mx, sd[k]);
    float ssum = 0.0f;
    for (int k = 0; k < 9; ++k) ssum += expf(sd[k] - mx);
    float w0 = expf(sd[0] - mx) / ssum;
    out_pred[b] = (1.0f - w0) * s_score;
  }
}

// ---------------- Kernel 5: bilinear upsample 28x28 -> 224x224 -------------------
__global__ __launch_bounds__(256) void k_upsample(const float* __restrict__ ps,
                                                  float* __restrict__ amap) {
  int idx = blockIdx.x * 256 + threadIdx.x;   // 8*224*224 = 401408 exact
  int b = idx / (OUTSZ * OUTSZ);
  int rem = idx % (OUTSZ * OUTSZ);
  int oy = rem / OUTSZ, ox = rem % OUTSZ;
  float iy = (oy + 0.5f) * (28.0f / 224.0f) - 0.5f;
  float ix = (ox + 0.5f) * (28.0f / 224.0f) - 0.5f;
  int y0 = (int)floorf(iy), x0 = (int)floorf(ix);
  float fy = iy - (float)y0, fx = ix - (float)x0;
  int y0c = iclampi(y0, 0, HW - 1), y1c = iclampi(y0 + 1, 0, HW - 1);
  int x0c = iclampi(x0, 0, HW - 1), x1c = iclampi(x0 + 1, 0, HW - 1);
  const float* p = ps + b * PPB;
  float v = (1.0f - fy) * ((1.0f - fx) * p[y0c * HW + x0c] + fx * p[y0c * HW + x1c])
          +         fy  * ((1.0f - fx) * p[y1c * HW + x0c] + fx * p[y1c * HW + x1c]);
  amap[idx] = v;
}

// ---------------- Kernel 6: separable Gaussian blur, reflect padding -------------
__global__ __launch_bounds__(256) void k_blur_h(const float* __restrict__ in,
                                                float* __restrict__ out) {
  __shared__ float kw[KSZ];
  if (threadIdx.x < KSZ) {
    float tt = (float)threadIdx.x - (float)KRAD;
    kw[threadIdx.x] = expf(-(tt * tt) / (2.0f * SIGMA * SIGMA));
  }
  __syncthreads();
  float ksum = 0.0f;
  for (int i = 0; i < KSZ; ++i) ksum += kw[i];
  int idx = blockIdx.x * 256 + threadIdx.x;  // 401408 exact
  int b = idx / (OUTSZ * OUTSZ);
  int rem = idx % (OUTSZ * OUTSZ);
  int y = rem / OUTSZ, x = rem % OUTSZ;
  const float* rowp = in + ((size_t)b * OUTSZ + y) * OUTSZ;
  float s = 0.0f;
  for (int i = 0; i < KSZ; ++i) {
    int xx = x + i - KRAD;
    xx = (xx < 0) ? -xx : ((xx > OUTSZ - 1) ? 2 * (OUTSZ - 1) - xx : xx);
    s += kw[i] * rowp[xx];
  }
  out[idx] = s / ksum;
}

__global__ __launch_bounds__(256) void k_blur_v(const float* __restrict__ in,
                                                float* __restrict__ out) {
  __shared__ float kw[KSZ];
  if (threadIdx.x < KSZ) {
    float tt = (float)threadIdx.x - (float)KRAD;
    kw[threadIdx.x] = expf(-(tt * tt) / (2.0f * SIGMA * SIGMA));
  }
  __syncthreads();
  float ksum = 0.0f;
  for (int i = 0; i < KSZ; ++i) ksum += kw[i];
  int idx = blockIdx.x * 256 + threadIdx.x;
  int b = idx / (OUTSZ * OUTSZ);
  int rem = idx % (OUTSZ * OUTSZ);
  int y = rem / OUTSZ, x = rem % OUTSZ;
  const float* img = in + (size_t)b * OUTSZ * OUTSZ;
  float s = 0.0f;
  for (int i = 0; i < KSZ; ++i) {
    int yy = y + i - KRAD;
    yy = (yy < 0) ? -yy : ((yy > OUTSZ - 1) ? 2 * (OUTSZ - 1) - yy : yy);
    s += kw[i] * img[yy * OUTSZ + x];
  }
  out[idx] = s / ksum;
}

// ---------------- Host orchestration ---------------------------------------------
extern "C" void kernel_launch(void* const* d_in, const int* in_sizes, int n_in,
                              void* d_out, int out_size, void* d_ws, size_t ws_size,
                              hipStream_t stream) {
  const float* feat2 = (const float*)d_in[0];   // [8,512,28,28]
  const float* feat3 = (const float*)d_in[1];   // [8,1024,14,14]
  const float* mbF   = (const float*)d_in[2];   // [20000,1536]

  char* ws = (char*)d_ws;
  size_t o = 0;
  auto take = [&](size_t bytes) -> char* {
    char* r = ws + o;
    o = (o + bytes + 255) & ~(size_t)255;
    return r;
  };
  bf16*  embB   = (bf16*) take((size_t)PTOT * EDIM * 2);
  float* embF   = (float*)take((size_t)PTOT * EDIM * 4);
  bf16*  mbB    = (bf16*) take((size_t)MROWS * EDIM * 2);
  float* ynorm  = (float*)take((size_t)MROWS * 4);
  float* xnorm  = (float*)take((size_t)PTOT * 4);
  float* f3p    = (float*)take((size_t)BATCH * C3 * H3 * H3 * 4);
  float* pscore = (float*)take((size_t)PTOT * 4);
  int*   ploc   = (int*)  take((size_t)PTOT * 4);
  float* dbank  = (float*)take((size_t)BATCH * MROWS * 4);
  float* amap   = (float*)take((size_t)BATCH * OUTSZ * OUTSZ * 4);
  float* tmp    = (float*)take((size_t)BATCH * OUTSZ * OUTSZ * 4);

  float* out_map  = (float*)d_out;                            // [8,1,224,224]
  float* out_pred = out_map + (size_t)BATCH * OUTSZ * OUTSZ;  // [8]

  k_pool3   <<<(BATCH * C3 * H3 * H3) / 256, 256, 0, stream>>>(feat3, f3p);
  k_embed   <<<(int)(((size_t)PTOT * EDIM) / 256), 256, 0, stream>>>(feat2, f3p, embF, embB);
  k_xnorm   <<<PTOT, 256, 0, stream>>>(embB, xnorm);
  k_mbprep  <<<MROWS, 256, 0, stream>>>(mbF, mbB, ynorm);
  k_gemm_min<<<PTOT / 16, 128, 0, stream>>>(embB, mbB, xnorm, ynorm, pscore, ploc);
  k_score   <<<BATCH, 256, 0, stream>>>(pscore, ploc, embF, mbF, dbank, out_pred);
  k_upsample<<<(BATCH * OUTSZ * OUTSZ) / 256, 256, 0, stream>>>(pscore, amap);
  k_blur_h  <<<(BATCH * OUTSZ * OUTSZ) / 256, 256, 0, stream>>>(amap, tmp);
  k_blur_v  <<<(BATCH * OUTSZ * OUTSZ) / 256, 256, 0, stream>>>(tmp, out_map);
}
